// MultiHeadAttention_30107720745670
// MI455X (gfx1250) — matrix operand order repaired and checked by reading, not compiled, not run on hardware
//
#include <hip/hip_runtime.h>

// Problem constants (fixed by the reference)
#define B_  2
#define S_  2048
#define D_  1024
#define H_  16
#define DK_ 64
#define DV_ 64
#define M_  (B_ * S_)        // 4096 rows in every GEMM
#define SCALE_ 0.125f        // DK^-0.5
#define NEGBIG (-3.30e38f)   // ~finfo(f32).min, finite so softmax-of-all-masked is uniform

typedef __bf16 bf16;
typedef __attribute__((ext_vector_type(16))) __bf16        v16bf;
typedef __attribute__((ext_vector_type(4)))  __bf16        v4bf;
typedef __attribute__((ext_vector_type(2)))  __bf16        v2bf;
typedef __attribute__((ext_vector_type(8)))  float         v8f;
typedef __attribute__((ext_vector_type(4)))  float         v4f;
typedef __attribute__((ext_vector_type(4)))  unsigned int  v4u;

union FragU { v16bf bf; v4u q[2]; unsigned w[8]; };
union PackBf2 { v2bf h; unsigned u; };
union Pack8Bf { v4u q; bf16 e[8]; };

static __device__ __forceinline__ v8f wmma_bf16(v16bf a, v16bf b, v8f c) {
  // D = A(16x32 bf16) * B(32x16 bf16) + C(16x16 f32)
  return __builtin_amdgcn_wmma_f32_16x16x32_bf16(false, a, false, b, (short)0, c,
                                                 false, false);
}

// ---------------------------------------------------------------------------
// f32 -> bf16 conversion (vectorized x4)
// ---------------------------------------------------------------------------
__global__ __launch_bounds__(256) void cvt_f32_to_bf16(const float* __restrict__ in,
                                                       bf16* __restrict__ out,
                                                       int nvec4) {
  int i = blockIdx.x * 256 + threadIdx.x;
  if (i >= nvec4) return;
  v4f x = ((const v4f*)in)[i];
  v4bf y;
  y[0] = (bf16)x[0]; y[1] = (bf16)x[1]; y[2] = (bf16)x[2]; y[3] = (bf16)x[3];
  ((v4bf*)out)[i] = y;
}

// ---------------------------------------------------------------------------
// WMMA GEMM:  out = A[M_ x 1024] @ Bc^T + bias,  Bc is [N x K] row-major
// (column n of the math-B is contiguous in K — matches W[n][k] layout).
// Each wave computes a 32x64 strip (2 row-subtiles x 4 col-tiles): B-frags are
// reused by both A rows -> 8 wmma per 12 b128 loads per K-step.
// MODE 0: bf16 out [B,H,S,DK] (heads)      — used for Q (scaled) and K
// MODE 1: bf16 out [B,H,DV,S] (transposed) — used for V
// MODE 2: f32  out [M_, D_] row-major      — final projection
// ---------------------------------------------------------------------------
template <int MODE>
__global__ __launch_bounds__(128) void gemm32x64_bf16(const bf16* __restrict__ A,
                                                      const bf16* __restrict__ Bc,
                                                      const float* __restrict__ bias,
                                                      void* __restrict__ outp,
                                                      float outScale) {
  const int lane  = threadIdx.x & 31;
  const int wid   = blockIdx.x * 4 + (threadIdx.x >> 5);
  const int nblks = D_ / 64;               // 16 column-strips
  const int m0    = (wid / nblks) * 32;
  const int n0    = (wid % nblks) * 64;
  const int ln    = lane & 15;
  const int half  = lane >> 4;

  const v8f vzero = {0.f, 0.f, 0.f, 0.f, 0.f, 0.f, 0.f, 0.f};
  v8f acc[2][4] = {{vzero, vzero, vzero, vzero}, {vzero, vzero, vzero, vzero}};

  const bf16* arow0 = A  + (size_t)(m0 + ln) * D_ + half * 16;       // A-frag sub 0
  const bf16* arow1 = A  + (size_t)(m0 + 16 + ln) * D_ + half * 16;  // A-frag sub 1
  const bf16* bcol  = Bc + (size_t)(n0 + ln) * D_ + half * 16;       // B-frag base

  for (int k0 = 0; k0 < D_; k0 += 32) {
    FragU fa0, fa1;
    fa0.q[0] = *(const v4u*)(arow0 + k0);
    fa0.q[1] = *(const v4u*)(arow0 + k0 + 8);
    fa1.q[0] = *(const v4u*)(arow1 + k0);
    fa1.q[1] = *(const v4u*)(arow1 + k0 + 8);
#pragma unroll
    for (int t = 0; t < 4; ++t) {
      FragU fb;
      const bf16* p = bcol + (size_t)t * 16 * D_ + k0;
      fb.q[0] = *(const v4u*)(p);
      fb.q[1] = *(const v4u*)(p + 8);
      acc[0][t] = wmma_bf16(fa0.bf, fb.bf, acc[0][t]);
      acc[1][t] = wmma_bf16(fa1.bf, fb.bf, acc[1][t]);
    }
  }

#pragma unroll
  for (int sub = 0; sub < 2; ++sub) {
#pragma unroll
    for (int t = 0; t < 4; ++t) {
      const int   ncol = n0 + t * 16 + ln;     // output column (lane-mapped)
      const float bb   = bias[ncol];
#pragma unroll
      for (int r = 0; r < 8; ++r) {
        const int   mrow = m0 + sub * 16 + r + 8 * half;  // output row (C-layout)
        const float v    = (acc[sub][t][r] + bb) * outScale;
        if (MODE == 0) {
          const int b = mrow >> 11, s = mrow & (S_ - 1);
          const int h = ncol >> 6,  d = ncol & 63;
          ((bf16*)outp)[(((size_t)(b * H_ + h)) * S_ + s) * DK_ + d] = (bf16)v;
        } else if (MODE == 1) {
          const int b = mrow >> 11, s = mrow & (S_ - 1);
          const int h = ncol >> 6,  d = ncol & 63;
          ((bf16*)outp)[(((size_t)(b * H_ + h)) * DV_ + d) * S_ + s] = (bf16)v;
        } else {
          ((float*)outp)[(size_t)mrow * D_ + ncol] = v;
        }
      }
    }
  }
}

// ---------------------------------------------------------------------------
// Flash attention, fully transposed so the softmax axis is in-lane.
// One wave per (b, h, 16-query strip); streams 32 keys/iteration.
//   scores^T = K_tile(16x64) @ Q^T(64x16)  -> lane owns one query column,
//              16 key-scores in registers.
//   O^T      = V^T(64x32) @ P^T(32x16)     -> alpha rescale is per-lane scalar;
//              P^T B-fragment built in registers via one xor-16 exchange.
// qp: [B*H, S, DK] bf16 (already scaled by DK^-0.5)
// kp: [B*H, S, DK] bf16
// vpT:[B*H, DV, S] bf16
// prev:[B*H, S, S] f32 (loaded as contiguous b128 runs of the lane's row)
// kpm:[B, S] bool(u8), oh:[B*S, H*DV] bf16
// ---------------------------------------------------------------------------
__global__ __launch_bounds__(128) void attn_wmma(const bf16* __restrict__ qp,
                                                 const bf16* __restrict__ kp,
                                                 const bf16* __restrict__ vpT,
                                                 const float* __restrict__ prev,
                                                 const unsigned char* __restrict__ kpm,
                                                 bf16* __restrict__ oh) {
  const int lane   = threadIdx.x & 31;
  const int wid    = blockIdx.x * 4 + (threadIdx.x >> 5);
  const int stiles = S_ / 16;                  // 128
  const int bh     = wid / stiles;
  const int s0     = (wid % stiles) * 16;
  const int b      = bh / H_;
  const int h      = bh % H_;
  const int ln     = lane & 15;
  const int half   = lane >> 4;

  const bf16*  qbase = qp  + (size_t)bh * S_ * DK_;
  const bf16*  kbase = kp  + (size_t)bh * S_ * DK_;
  const bf16*  vbase = vpT + (size_t)bh * DV_ * S_;
  const float* prow  = prev + (size_t)bh * S_ * S_ + (size_t)(s0 + ln) * S_;
  const unsigned char* mbase = kpm + (size_t)b * S_;

  // Q^T B-fragments for the full DK=64 (two K=32 steps), resident all loop.
  FragU bq0, bq1;
  {
    const bf16* p = qbase + (size_t)(s0 + ln) * DK_ + half * 16;
    bq0.q[0] = *(const v4u*)(p);
    bq0.q[1] = *(const v4u*)(p + 8);
    bq1.q[0] = *(const v4u*)(p + 32);
    bq1.q[1] = *(const v4u*)(p + 40);
  }

  const v8f vzero = {0.f, 0.f, 0.f, 0.f, 0.f, 0.f, 0.f, 0.f};
  v8f acc[4] = {vzero, vzero, vzero, vzero};   // O^T tiles: dv rows x 16 queries
  float mQ = -__builtin_inff();                // per-lane (per-query) running max
  float lQ = 0.f;                              // per-lane running denominator

  for (int t0 = 0; t0 < S_; t0 += 32) {
    // ---- scores^T: two 16x16 tiles (keys t0..+15 as rows, then t0+16..+31) --
    v8f clo = vzero, chi = vzero;
    {
      FragU ak;
      const bf16* p = kbase + (size_t)(t0 + ln) * DK_ + half * 16;
      ak.q[0] = *(const v4u*)(p);       ak.q[1] = *(const v4u*)(p + 8);
      clo = wmma_bf16(ak.bf, bq0.bf, clo);
      ak.q[0] = *(const v4u*)(p + 32);  ak.q[1] = *(const v4u*)(p + 40);
      clo = wmma_bf16(ak.bf, bq1.bf, clo);
      const bf16* p2 = kbase + (size_t)(t0 + 16 + ln) * DK_ + half * 16;
      ak.q[0] = *(const v4u*)(p2);      ak.q[1] = *(const v4u*)(p2 + 8);
      chi = wmma_bf16(ak.bf, bq0.bf, chi);
      ak.q[0] = *(const v4u*)(p2 + 32); ak.q[1] = *(const v4u*)(p2 + 40);
      chi = wmma_bf16(ak.bf, bq1.bf, chi);
    }

    // prefetch next prev run (dominant HBM stream; speculative, OOB-safe)
    __builtin_prefetch(prow + t0 + 32, 0, 0);

    // ---- prev residual: contiguous b128 runs of this lane's query row ------
    // clo[r] ~ key t0 + 8*half + r ; chi[r] ~ key t0 + 16 + 8*half + r
    v4f pv0 = *(const v4f*)(prow + t0 + 8 * half);
    v4f pv1 = *(const v4f*)(prow + t0 + 8 * half + 4);
    v4f pv2 = *(const v4f*)(prow + t0 + 16 + 8 * half);
    v4f pv3 = *(const v4f*)(prow + t0 + 16 + 8 * half + 4);

    // ---- key padding mask: 8 contiguous bool bytes per tile ---------------
    const unsigned long long mlo =
        *(const unsigned long long*)(mbase + t0 + 8 * half);
    const unsigned long long mhi =
        *(const unsigned long long*)(mbase + t0 + 16 + 8 * half);

    float sl[8], sh[8];
#pragma unroll
    for (int r = 0; r < 8; ++r) {
      const float pl = (r < 4) ? pv0[r & 3] : pv1[r & 3];
      const float ph = (r < 4) ? pv2[r & 3] : pv3[r & 3];
      const bool  bl = ((mlo >> (8 * r)) & 0xffull) != 0ull;
      const bool  bh2 = ((mhi >> (8 * r)) & 0xffull) != 0ull;
      sl[r] = bl  ? NEGBIG : (clo[r] + pl);
      sh[r] = bh2 ? NEGBIG : (chi[r] + ph);
    }

    // ---- online softmax: in-lane reduction + one xor-16 combine -----------
    float tmax = fmaxf(sl[0], sh[0]);
#pragma unroll
    for (int r = 1; r < 8; ++r) tmax = fmaxf(tmax, fmaxf(sl[r], sh[r]));
    tmax = fmaxf(tmax, __shfl_xor(tmax, 16));

    const float mnew  = fmaxf(mQ, tmax);
    const float alpha = __expf(mQ - mnew);

    float pl[8], ph[8], rs = 0.f;
#pragma unroll
    for (int r = 0; r < 8; ++r) {
      pl[r] = __expf(sl[r] - mnew);
      ph[r] = __expf(sh[r] - mnew);
      rs += pl[r] + ph[r];
    }
    rs += __shfl_xor(rs, 16);

    lQ = lQ * alpha + rs;
    mQ = mnew;
#pragma unroll
    for (int t = 0; t < 4; ++t) acc[t] = acc[t] * alpha;

    // ---- build P^T B-fragment in registers (keys K = 16*half + j) ---------
    // own 8 p-values + 8 from the xor-16 partner, exchanged as packed bf16.
    unsigned plo_pk[4], phi_pk[4];
#pragma unroll
    for (int i = 0; i < 4; ++i) {
      PackBf2 a, c;
      a.h[0] = (bf16)pl[2 * i]; a.h[1] = (bf16)pl[2 * i + 1];
      c.h[0] = (bf16)ph[2 * i]; c.h[1] = (bf16)ph[2 * i + 1];
      plo_pk[i] = a.u; phi_pk[i] = c.u;
    }
    FragU bP;
#pragma unroll
    for (int i = 0; i < 4; ++i) {
      const unsigned send = half ? plo_pk[i] : phi_pk[i];
      const unsigned recv = __shfl_xor(send, 16);
      bP.w[i]     = half ? recv      : plo_pk[i];  // elems j=0..7  : keys 16h+0..7
      bP.w[4 + i] = half ? phi_pk[i] : recv;       // elems j=8..15 : keys 16h+8..15
    }

    // ---- O^T += V^T(16dv x 32keys) @ P^T ----------------------------------
#pragma unroll
    for (int t = 0; t < 4; ++t) {
      FragU av;
      const bf16* p = vbase + (size_t)(t * 16 + ln) * S_ + t0 + half * 16;
      av.q[0] = *(const v4u*)(p);
      av.q[1] = *(const v4u*)(p + 8);
      acc[t] = wmma_bf16(av.bf, bP.bf, acc[t]);
    }
  }

  // ---- normalize, pack 8 consecutive bf16, one b128 store per tile --------
  const float inv = 1.0f / lQ;
  bf16* orow = oh + (size_t)(b * S_ + s0 + ln) * (H_ * DV_) + h * DV_;
#pragma unroll
  for (int t = 0; t < 4; ++t) {
    Pack8Bf pk;
#pragma unroll
    for (int r = 0; r < 8; ++r) pk.e[r] = (bf16)(acc[t][r] * inv);
    *(v4u*)(orow + t * 16 + 8 * half) = pk.q;   // dv cols 16t + 8*half + [0..7]
  }
}

// ---------------------------------------------------------------------------
// Host side
// ---------------------------------------------------------------------------
extern "C" void kernel_launch(void* const* d_in, const int* in_sizes, int n_in,
                              void* d_out, int out_size, void* d_ws, size_t ws_size,
                              hipStream_t stream) {
  (void)in_sizes; (void)n_in; (void)out_size; (void)ws_size;

  const float* q    = (const float*)d_in[0];
  const float* k    = (const float*)d_in[1];
  const float* v    = (const float*)d_in[2];
  const float* prev = (const float*)d_in[3];
  const unsigned char* mask = (const unsigned char*)d_in[4];  // jnp.bool_ = 1 byte
  const float* Wq = (const float*)d_in[5];
  const float* bq = (const float*)d_in[6];
  const float* Wk = (const float*)d_in[7];
  const float* bk = (const float*)d_in[8];
  const float* Wv = (const float*)d_in[9];
  const float* bv = (const float*)d_in[10];
  const float* Wo = (const float*)d_in[11];
  const float* bo = (const float*)d_in[12];
  float* out = (float*)d_out;

  // Workspace carve-up (all sizes are large powers of two -> aligned). 64 MB.
  char*  w = (char*)d_ws;
  size_t o = 0;
  auto take = [&](size_t bytes) { char* p = w + o; o += bytes; return (void*)p; };
  bf16* qb   = (bf16*)take((size_t)M_ * D_ * 2);             // 8 MB
  bf16* kb   = (bf16*)take((size_t)M_ * D_ * 2);
  bf16* vb   = (bf16*)take((size_t)M_ * D_ * 2);
  bf16* Wqb  = (bf16*)take((size_t)D_ * D_ * 2);             // 2 MB
  bf16* Wkb  = (bf16*)take((size_t)D_ * D_ * 2);
  bf16* Wvb  = (bf16*)take((size_t)D_ * D_ * 2);
  bf16* Wob  = (bf16*)take((size_t)D_ * D_ * 2);
  bf16* qpj  = (bf16*)take((size_t)B_ * H_ * S_ * DK_ * 2);  // 8 MB
  bf16* kpj  = (bf16*)take((size_t)B_ * H_ * S_ * DK_ * 2);
  bf16* vpT  = (bf16*)take((size_t)B_ * H_ * DV_ * S_ * 2);
  bf16* oh   = (bf16*)take((size_t)M_ * (H_ * DV_) * 2);     // 8 MB

  // 1) f32 -> bf16 conversions
  const int nvAct = M_ * D_ / 4;   // 1,048,576 vec4 -> 4096 blocks
  const int nvW   = D_ * D_ / 4;   //   262,144 vec4 -> 1024 blocks
  cvt_f32_to_bf16<<<nvAct / 256, 256, 0, stream>>>(q,  qb,  nvAct);
  cvt_f32_to_bf16<<<nvAct / 256, 256, 0, stream>>>(k,  kb,  nvAct);
  cvt_f32_to_bf16<<<nvAct / 256, 256, 0, stream>>>(v,  vb,  nvAct);
  cvt_f32_to_bf16<<<nvW / 256,   256, 0, stream>>>(Wq, Wqb, nvW);
  cvt_f32_to_bf16<<<nvW / 256,   256, 0, stream>>>(Wk, Wkb, nvW);
  cvt_f32_to_bf16<<<nvW / 256,   256, 0, stream>>>(Wv, Wvb, nvW);
  cvt_f32_to_bf16<<<nvW / 256,   256, 0, stream>>>(Wo, Wob, nvW);

  // 2) QKV projections: 2048 waves = 512 blocks x 4 waves each
  const int gBlocks = (M_ / 32) * (D_ / 64) / 4;  // 512
  gemm32x64_bf16<0><<<gBlocks, 128, 0, stream>>>(qb, Wqb, bq, qpj, SCALE_);
  gemm32x64_bf16<0><<<gBlocks, 128, 0, stream>>>(kb, Wkb, bk, kpj, 1.0f);
  gemm32x64_bf16<1><<<gBlocks, 128, 0, stream>>>(vb, Wvb, bv, vpT, 1.0f);

  // 3) attention: B*H*(S/16) = 4096 waves = 1024 blocks
  const int aBlocks = B_ * H_ * (S_ / 16) / 4;    // 1024
  attn_wmma<<<aBlocks, 128, 0, stream>>>(qpj, kpj, vpT, prev, mask, oh);

  // 4) output projection -> f32 d_out [B*S, D]
  gemm32x64_bf16<2><<<gBlocks, 128, 0, stream>>>(oh, Wob, bo, (void*)out, 1.0f);
}
